// MultiHeadAttention_11493332484097
// MI455X (gfx1250) — matrix operand order repaired and checked by reading, not compiled, not run on hardware
//
#include <hip/hip_runtime.h>
#include <hip/hip_bf16.h>

#define DMODEL 2048
#define NHEADS 16
#define DK     128
#define BATCH  2
#define SEQ    2048
#define ROWS   (BATCH * SEQ)

typedef __attribute__((ext_vector_type(16))) _Float16 v16h;
typedef __attribute__((ext_vector_type(8)))  _Float16 v8h;
typedef __attribute__((ext_vector_type(8)))  float    v8f;

__device__ __forceinline__ v8f wmma_f16(v16h a, v16h b, v8f c) {
  // D = A(16x32 f16) * B(32x16 f16) + C(16x16 f32)
  return __builtin_amdgcn_wmma_f32_16x16x32_f16(false, a, false, b, (short)0, c,
                                                false, false);
}

__device__ __forceinline__ v16h cat8(v8h lo, v8h hi) {
  return __builtin_shufflevector(lo, hi, 0, 1, 2, 3, 4, 5, 6, 7, 8, 9, 10, 11,
                                 12, 13, 14, 15);
}

// CDNA5 async global->LDS copy (ASYNCcnt-tracked), 16 bytes per lane.
// LDS address comes from the vdst VGPR; low 32 bits of a generic shared pointer
// are the LDS offset (ISA aperture rule).
__device__ __forceinline__ void async_copy_b128(_Float16* lds_dst,
                                                const _Float16* gsrc) {
  uint32_t lds_off = (uint32_t)(uintptr_t)lds_dst;
  asm volatile("global_load_async_to_lds_b128 %0, %1, off"
               :
               : "v"(lds_off), "v"(gsrc)
               : "memory");
}
// ASYNCcnt is per-wave, in-order: waiting <= N (N = size of the batch just
// issued) drains exactly the previous batch while the new one stays in flight.
__device__ __forceinline__ void wait_async0() {
  asm volatile("s_wait_asynccnt 0x0" ::: "memory");
}
__device__ __forceinline__ void wait_async4() {
  asm volatile("s_wait_asynccnt 0x4" ::: "memory");
}
__device__ __forceinline__ void wait_async8() {
  asm volatile("s_wait_asynccnt 0x8" ::: "memory");
}

// ---------------------------------------------------------------- prep ----
__global__ void cast_x_f16(const float* __restrict__ x,
                           _Float16* __restrict__ xh, int n) {
  int i = blockIdx.x * blockDim.x + threadIdx.x;
  if (i < n) xh[i] = (_Float16)x[i];
}

// wt[k*D + n] = (f16) w[n*D + k]   (so WMMA B-frag loads are contiguous)
__global__ void transpose_w_f16(const float* __restrict__ w0,
                                const float* __restrict__ w1,
                                const float* __restrict__ w2,
                                const float* __restrict__ w3,
                                _Float16* __restrict__ t0,
                                _Float16* __restrict__ t1,
                                _Float16* __restrict__ t2,
                                _Float16* __restrict__ t3) {
  __shared__ float tile[16][17];
  const float* w;
  _Float16* t;
  switch (blockIdx.z) {
    case 0:  w = w0; t = t0; break;
    case 1:  w = w1; t = t1; break;
    case 2:  w = w2; t = t2; break;
    default: w = w3; t = t3; break;
  }
  int tx = threadIdx.x & 15, ty = threadIdx.x >> 4;
  tile[ty][tx] = w[(size_t)(blockIdx.y * 16 + ty) * DMODEL + blockIdx.x * 16 + tx];
  __syncthreads();
  t[(size_t)(blockIdx.x * 16 + ty) * DMODEL + blockIdx.y * 16 + tx] =
      (_Float16)tile[tx][ty];
}

// -------------------------------------------- QKV projection + RoPE ------
// 256 threads (8 waves) compute a 128x128 tile of y = x @ W^T.
// K=2048 in 64-chunks, double-buffered LDS (A 128x64, B 64x128 per buffer),
// async global->LDS prefetch one chunk ahead.  Wave w owns rows
// [w*16,w*16+16) x 8 col tiles: 16 WMMA per chunk per wave.
// Epilogue applies RoPE (q,k) via lane^1 shuffle and scatters f16 into
// attention layouts (K transposed: (B,H,Dk,S)).
__global__ __launch_bounds__(256) void gemm_qkv_rope(
    const _Float16* __restrict__ xh, const _Float16* __restrict__ wtq,
    const _Float16* __restrict__ wtk, const _Float16* __restrict__ wtv,
    _Float16* __restrict__ qh, _Float16* __restrict__ kht,
    _Float16* __restrict__ vh) {
  __shared__ __align__(64) _Float16 sA[2][128 * 64];  // [buf][row][k]
  __shared__ __align__(64) _Float16 sB[2][64 * 128];  // [buf][k][n]
  const int tid = threadIdx.x;
  const int wave = tid >> 5, lane = tid & 31;
  const int m = lane & 15, khalf = lane >> 4;
  const int rt0 = blockIdx.x * 128, nt0 = blockIdx.y * 128;
  const int which = blockIdx.z;
  const _Float16* Bm = (which == 0) ? wtq : ((which == 1) ? wtk : wtv);

  // stage one 64-wide K chunk: 8 async b128 per thread (4 A + 4 B)
  auto stage = [&](int kc, int buf) {
#pragma unroll
    for (int i = 0; i < 4; ++i) {
      int s = tid + i * 256;           // 1024 16B segments each
      int ar = s >> 3, aq = s & 7;     // A: row, 8-elem chunk of 64
      async_copy_b128(&sA[buf][ar * 64 + aq * 8],
                      xh + (size_t)(rt0 + ar) * DMODEL + kc + aq * 8);
      int bk = s >> 4, bn = s & 15;    // B: k-row, 8-elem chunk of 128
      async_copy_b128(&sB[buf][bk * 128 + bn * 8],
                      Bm + (size_t)(kc + bk) * DMODEL + nt0 + bn * 8);
    }
  };

  const v8f vzero = {0.f, 0.f, 0.f, 0.f, 0.f, 0.f, 0.f, 0.f};
  v8f acc[8];
#pragma unroll
  for (int i = 0; i < 8; ++i) acc[i] = vzero;

  const int NC = DMODEL / 64;
  stage(0, 0);
  for (int ic = 0; ic < NC; ++ic) {
    const int cur = ic & 1;
    __syncthreads();  // all readers of buffer cur^1 (iter ic-1) are done
    if (ic + 1 < NC) {
      stage((ic + 1) * 64, cur ^ 1);
      wait_async8();  // drain chunk ic's batch; chunk ic+1 stays in flight
    } else {
      wait_async0();
    }
    __syncthreads();  // chunk ic visible to all waves

#pragma unroll
    for (int hh = 0; hh < 2; ++hh) {  // two 32-wide K sub-chunks
      v8h alo = *(const v8h*)(&sA[cur][(wave * 16 + m) * 64 + hh * 32 + khalf * 8]);
      v8h ahi = *(const v8h*)(&sA[cur][(wave * 16 + m) * 64 + hh * 32 + 16 + khalf * 8]);
      v16h af = cat8(alo, ahi);
#pragma unroll
      for (int nt2 = 0; nt2 < 8; ++nt2) {
        v16h bf;
#pragma unroll
        for (int s = 0; s < 16; ++s)
          bf[s] = sB[cur][(hh * 32 + khalf * 8 + (s & 7) + ((s >> 3) << 4)) * 128 + nt2 * 16 + m];
        acc[nt2] = wmma_f16(af, bf, acc[nt2]);
      }
    }
  }

  // C layout: elem v -> row M = v + 8*khalf, col n = lane&15
#pragma unroll
  for (int nt2 = 0; nt2 < 8; ++nt2) {
#pragma unroll
    for (int v = 0; v < 8; ++v) {
      int row = rt0 + wave * 16 + v + 8 * khalf;  // token row in [0, B*S)
      int bb = row / SEQ, ss = row % SEQ;
      int f = nt0 + nt2 * 16 + m;  // feature in [0, D)
      int h = f >> 7, d = f & 127;
      float val = acc[nt2][v];
      float pv = __shfl_xor(val, 1, 32);  // rotation partner (d ^ 1)
      if (which < 2) {                    // RoPE on q and k
        float i2 = (float)((d >> 1) * 2);
        float ang = (float)ss * __powf(10000.f, -i2 * (1.0f / 128.0f));
        float sn, cs;
        __sincosf(ang, &sn, &cs);
        val = (d & 1) ? (pv * sn + val * cs) : (val * cs - pv * sn);
      }
      size_t bhh = (size_t)bb * NHEADS + h;
      if (which == 0)
        qh[(bhh * SEQ + ss) * DK + d] = (_Float16)val;
      else if (which == 1)
        kht[(bhh * DK + d) * SEQ + ss] = (_Float16)val;  // transposed
      else
        vh[(bhh * SEQ + ss) * DK + d] = (_Float16)val;
    }
  }
}

// ------------------------------------------------ causal flash attention --
// 256 threads (8 waves) process 128 query rows of one (b,h); wave w owns
// q-tile qt0+w.  32-key K/V tiles staged in double-buffered LDS with async
// prefetch one chunk ahead, shared by all 8 waves.  Per 32 keys: 8 WMMA for
// S = Q.K^T, online softmax, P routed C->A layout through each wave's
// private LDS region, 8 WMMA for O += P.V.
__global__ __launch_bounds__(256) void flash_attn(
    const _Float16* __restrict__ qh, const _Float16* __restrict__ kht,
    const _Float16* __restrict__ vh, _Float16* __restrict__ oh) {
  __shared__ __align__(64) _Float16 sK[2][128 * 32];    // [buf][dk][key]
  __shared__ __align__(64) _Float16 sV[2][32 * 128];    // [buf][key][dk]
  __shared__ __align__(64) _Float16 sP[8 * 16 * 32];    // per-wave P buffers
  const int tid = threadIdx.x;
  const int wave = tid >> 5, lane = tid & 31;
  const int m = lane & 15, khalf = lane >> 4;
  const int qt0 = blockIdx.x * 8, bh = blockIdx.y;
  const int qt = qt0 + wave;
  const int b = bh / NHEADS, h = bh % NHEADS;
  const float sm = 0.08838834764831845f;  // 1/sqrt(128)
  _Float16* pb = sP + wave * (16 * 32);

  auto stage = [&](int kv, int buf) {
#pragma unroll
    for (int i = 0; i < 2; ++i) {
      int s = tid + i * 256;           // 512 16B segments each
      int dk = s >> 2, kq = s & 3;
      async_copy_b128(&sK[buf][dk * 32 + kq * 8],
                      kht + ((size_t)bh * DK + dk) * SEQ + kv + kq * 8);
      int key = s >> 4, n8 = s & 15;
      async_copy_b128(&sV[buf][key * 128 + n8 * 8],
                      vh + ((size_t)bh * SEQ + kv + key) * DK + n8 * 8);
    }
  };

  // Q A-fragments: 4 chunks covering Dk=128 (per-wave rows, from global)
  v16h qf[4];
  const _Float16* qrow = qh + ((size_t)bh * SEQ + qt * 16 + m) * DK;
#pragma unroll
  for (int c = 0; c < 4; ++c) {
    v8h lo = *(const v8h*)(qrow + c * 32 + khalf * 8);
    v8h hi = *(const v8h*)(qrow + c * 32 + 16 + khalf * 8);
    qf[c] = cat8(lo, hi);
  }

  const v8f vzero = {0.f, 0.f, 0.f, 0.f, 0.f, 0.f, 0.f, 0.f};
  v8f o[8];
  float row_m[8], row_l[8];
#pragma unroll
  for (int i = 0; i < 8; ++i) {
    o[i] = vzero;
    row_m[i] = -1e30f;
    row_l[i] = 0.f;
  }

  const int kvmax = (qt0 + 7) * 16;  // last wave's diagonal
  stage(0, 0);
  for (int kv = 0, ic = 0; kv <= kvmax; kv += 32, ++ic) {
    const int cur = ic & 1;
    __syncthreads();  // readers of buffer cur^1 (iter ic-1) done
    if (kv + 32 <= kvmax) {
      stage(kv + 32, cur ^ 1);
      wait_async4();  // drain this chunk's batch; next stays in flight
    } else {
      wait_async0();
    }
    __syncthreads();  // chunk visible to all waves

    if (kv <= qt * 16) {  // wave-uniform causal gating
      v8f s0 = vzero, s1 = vzero;
#pragma unroll
      for (int c = 0; c < 4; ++c) {
        const _Float16* kb = &sK[cur][(c * 32) * 32 + m];  // lane = key col, elems = dk
        v16h b0, b1;
#pragma unroll
        for (int s = 0; s < 16; ++s) {
          int kk = khalf * 8 + (s & 7) + ((s >> 3) << 4);
          b0[s] = kb[kk * 32];
          b1[s] = kb[kk * 32 + 16];
        }
        s0 = wmma_f16(qf[c], b0, s0);
        s1 = wmma_f16(qf[c], b1, s1);
      }

      float p0[8], p1[8], alp[8];
#pragma unroll
      for (int v = 0; v < 8; ++v) {
        int qi = qt * 16 + v + 8 * khalf;
        int j0 = kv + m, j1 = kv + 16 + m;
        float a0 = (j0 <= qi) ? s0[v] * sm : -1e30f;  // causal mask
        float a1 = (j1 <= qi) ? s1[v] * sm : -1e30f;
        float mx = fmaxf(a0, a1);
#pragma unroll
        for (int msk = 1; msk < 16; msk <<= 1)
          mx = fmaxf(mx, __shfl_xor(mx, msk, 32));  // row max (16 lanes)
        float mnew = fmaxf(row_m[v], mx);
        float e0 = __expf(a0 - mnew);
        float e1 = __expf(a1 - mnew);
        float al = __expf(row_m[v] - mnew);
        float rs = e0 + e1;
#pragma unroll
        for (int msk = 1; msk < 16; msk <<= 1) rs += __shfl_xor(rs, msk, 32);
        row_m[v] = mnew;
        row_l[v] = row_l[v] * al + rs;
        p0[v] = e0;
        p1[v] = e1;
        alp[v] = al;
      }
#pragma unroll
      for (int nt2 = 0; nt2 < 8; ++nt2)
#pragma unroll
        for (int v = 0; v < 8; ++v) o[nt2][v] *= alp[v];

      // P: C layout -> per-wave LDS (16x32 row-major) -> A layout
      // (LDS ops are in-order within a wave; private region, no barrier)
#pragma unroll
      for (int v = 0; v < 8; ++v) {
        int M = v + 8 * khalf;
        pb[M * 32 + m] = (_Float16)p0[v];
        pb[M * 32 + 16 + m] = (_Float16)p1[v];
      }
      v8h plo = *(const v8h*)(pb + m * 32 + khalf * 8);
      v8h phi = *(const v8h*)(pb + m * 32 + 16 + khalf * 8);
      v16h pf = cat8(plo, phi);

      // O += P.V : lane = dk col, elems = keys in table order (from LDS)
#pragma unroll
      for (int nt2 = 0; nt2 < 8; ++nt2) {
        v16h vf;
#pragma unroll
        for (int s = 0; s < 16; ++s)
          vf[s] = sV[cur][(khalf * 8 + (s & 7) + ((s >> 3) << 4)) * 128 + nt2 * 16 + m];
        o[nt2] = wmma_f16(pf, vf, o[nt2]);
      }
    }
  }

  // normalize and write (B, S, D) f16 with D index = h*128 + dk
#pragma unroll
  for (int nt2 = 0; nt2 < 8; ++nt2) {
#pragma unroll
    for (int v = 0; v < 8; ++v) {
      int qi = qt * 16 + v + 8 * khalf;
      float val = o[nt2][v] / row_l[v];
      oh[((size_t)b * SEQ + qi) * DMODEL + h * DK + nt2 * 16 + m] =
          (_Float16)val;
    }
  }
}

// ------------------------------------------------- output projection -----
// Same double-buffered macro-tiled structure; epilogue adds bias, fp32 out.
__global__ __launch_bounds__(256) void gemm_out(const _Float16* __restrict__ oh,
                                                const _Float16* __restrict__ wto,
                                                const float* __restrict__ bo,
                                                float* __restrict__ y) {
  __shared__ __align__(64) _Float16 sA[2][128 * 64];
  __shared__ __align__(64) _Float16 sB[2][64 * 128];
  const int tid = threadIdx.x;
  const int wave = tid >> 5, lane = tid & 31;
  const int m = lane & 15, khalf = lane >> 4;
  const int rt0 = blockIdx.x * 128, nt0 = blockIdx.y * 128;

  auto stage = [&](int kc, int buf) {
#pragma unroll
    for (int i = 0; i < 4; ++i) {
      int s = tid + i * 256;
      int ar = s >> 3, aq = s & 7;
      async_copy_b128(&sA[buf][ar * 64 + aq * 8],
                      oh + (size_t)(rt0 + ar) * DMODEL + kc + aq * 8);
      int bk = s >> 4, bn = s & 15;
      async_copy_b128(&sB[buf][bk * 128 + bn * 8],
                      wto + (size_t)(kc + bk) * DMODEL + nt0 + bn * 8);
    }
  };

  const v8f vzero = {0.f, 0.f, 0.f, 0.f, 0.f, 0.f, 0.f, 0.f};
  v8f acc[8];
#pragma unroll
  for (int i = 0; i < 8; ++i) acc[i] = vzero;

  const int NC = DMODEL / 64;
  stage(0, 0);
  for (int ic = 0; ic < NC; ++ic) {
    const int cur = ic & 1;
    __syncthreads();
    if (ic + 1 < NC) {
      stage((ic + 1) * 64, cur ^ 1);
      wait_async8();
    } else {
      wait_async0();
    }
    __syncthreads();

#pragma unroll
    for (int hh = 0; hh < 2; ++hh) {
      v8h alo = *(const v8h*)(&sA[cur][(wave * 16 + m) * 64 + hh * 32 + khalf * 8]);
      v8h ahi = *(const v8h*)(&sA[cur][(wave * 16 + m) * 64 + hh * 32 + 16 + khalf * 8]);
      v16h af = cat8(alo, ahi);
#pragma unroll
      for (int nt2 = 0; nt2 < 8; ++nt2) {
        v16h bf;
#pragma unroll
        for (int s = 0; s < 16; ++s)
          bf[s] = sB[cur][(hh * 32 + khalf * 8 + (s & 7) + ((s >> 3) << 4)) * 128 + nt2 * 16 + m];
        acc[nt2] = wmma_f16(af, bf, acc[nt2]);
      }
    }
  }

#pragma unroll
  for (int nt2 = 0; nt2 < 8; ++nt2) {
    int f = nt0 + nt2 * 16 + m;
    float bias = bo[f];
#pragma unroll
    for (int v = 0; v < 8; ++v) {
      int row = rt0 + wave * 16 + v + 8 * khalf;
      y[(size_t)row * DMODEL + f] = acc[nt2][v] + bias;
    }
  }
}

// -------------------------------------------------------------- launch ---
extern "C" void kernel_launch(void* const* d_in, const int* in_sizes, int n_in,
                              void* d_out, int out_size, void* d_ws,
                              size_t ws_size, hipStream_t stream) {
  const float* x = (const float*)d_in[0];
  const float* wq = (const float*)d_in[1];
  const float* wk = (const float*)d_in[2];
  const float* wv = (const float*)d_in[3];
  const float* wo = (const float*)d_in[4];
  const float* bo = (const float*)d_in[5];
  float* y = (float*)d_out;

  char* ws = (char*)d_ws;
  const size_t XH = (size_t)ROWS * DMODEL * sizeof(_Float16);
  const size_t WB = (size_t)DMODEL * DMODEL * sizeof(_Float16);
  const size_t QK = XH;
  const size_t PAD = 8192;  // diagonal-tile overreach of last row

  size_t off = 0;
  _Float16* xh = (_Float16*)(ws + off);  off += XH;
  _Float16* wtq = (_Float16*)(ws + off); off += WB;
  _Float16* wtk = (_Float16*)(ws + off); off += WB;
  _Float16* wtv = (_Float16*)(ws + off); off += WB;
  _Float16* wto = (_Float16*)(ws + off); off += WB;
  _Float16* qh = (_Float16*)(ws + off);  off += QK;
  _Float16* kht = (_Float16*)(ws + off); off += QK + PAD;
  _Float16* vh = (_Float16*)(ws + off);  off += QK + PAD;
  _Float16* oh = xh;  // xh dead after gemm_qkv_rope

  const int nx = ROWS * DMODEL;
  cast_x_f16<<<(nx + 255) / 256, 256, 0, stream>>>(x, xh, nx);
  transpose_w_f16<<<dim3(DMODEL / 16, DMODEL / 16, 4), 256, 0, stream>>>(
      wq, wk, wv, wo, wtq, wtk, wtv, wto);
  gemm_qkv_rope<<<dim3(ROWS / 128, DMODEL / 128, 3), 256, 0, stream>>>(
      xh, wtq, wtk, wtv, qh, kht, vh);
  flash_attn<<<dim3(SEQ / 128, BATCH * NHEADS), 256, 0, stream>>>(qh, kht, vh,
                                                                  oh);
  gemm_out<<<dim3(ROWS / 128, DMODEL / 128), 256, 0, stream>>>(oh, wto, bo, y);
}
